// ReAttention_18373870092683
// MI455X (gfx1250) — compile-verified, matrix operand order checked by reading, not run on hardware
//
#include <hip/hip_runtime.h>
#include <math.h>
#include <stdint.h>

// ---------------- problem constants ----------------
#define BX 8
#define NX 1024
#define CX 768
#define HX 12
#define DX 64
#define SCALE_Q 0.125f
#define BN_EPS 1e-5f

static const long XB_E    = (long)BX * NX * CX;        // 6291456  x (f16 copy)
static const long WQKV_E  = 3L * CX * CX;              // 1769472
static const long WPROJ_E = (long)CX * CX;             // 589824
static const long HEAD_E  = (long)BX * HX * NX * DX;   // 6291456  per q/k/v
static const long OUT_E   = (long)BX * NX * CX;        // 6291456  final out
static const long ATTN_E  = (long)BX * HX * NX * NX;   // 100663296

// ---------------- WMMA helpers (wave32, 16x16x32 f16 -> f32) ----------------
typedef __attribute__((ext_vector_type(16))) _Float16 v16h;
typedef __attribute__((ext_vector_type(8)))  float    v8f;

__device__ __forceinline__ v8f wmma32(v16h a, v16h b, v8f c) {
  return __builtin_amdgcn_wmma_f32_16x16x32_f16(false, a, false, b, (short)0, c,
                                                false, false);
}

// Fragment loader for an operand tile stored ROW-MAJOR [row][k] in LDS
// (A: row = M row; B: store B transposed [n][k] and row = N column).
// lane L: row = L&15, half g = L>>4; elems are two contiguous 8-half runs:
// k in [8g, 8g+8) and [8g+16, 8g+24)  ->  two ds_load_b128.
__device__ __forceinline__ v16h frag_rm(const _Float16* p, int ld) {
  const int lane = threadIdx.x & 31;
  const int row = lane & 15, g = lane >> 4;
  const _Float16* q = p + row * ld + 8 * g;
  union { v16h v; uint4 u[2]; } f;
  f.u[0] = *(const uint4*)(q);
  f.u[1] = *(const uint4*)(q + 16);
  return f.v;
}
// C/D fragment element (r, lane): row = r + 8*(lane>>4), col = lane&15.

// ---------------- CDNA5 async memory->LDS (ASYNCcnt-tracked) ----------------
// Per ISA 10.2: generic LDS address low 32 bits are the LDS byte offset.
__device__ __forceinline__ void async_copy_b128(void* lds_ptr, const void* gptr) {
  uint32_t loff = (uint32_t)(uintptr_t)lds_ptr;
  unsigned long long ga = (unsigned long long)(uintptr_t)gptr;
  asm volatile("global_load_async_to_lds_b128 %0, %1, off"
               :: "v"(loff), "v"(ga) : "memory");
}
__device__ __forceinline__ void async_wait0() {
  asm volatile("s_wait_asynccnt 0x0" ::: "memory");
}

// ---------------- fp32 -> f16 convert ----------------
__global__ void cvt_f32_f16(const float* __restrict__ in, _Float16* __restrict__ out,
                            long n) {
  long i = (long)blockIdx.x * blockDim.x + threadIdx.x;
  if (i < n) out[i] = (_Float16)in[i];
}

// ---------------- QKV GEMM: [8192,768] x [2304,768]^T, scatter q/k/vT ----------------
__global__ __launch_bounds__(256) void qkv_gemm(const _Float16* __restrict__ xb,
                                                const _Float16* __restrict__ wb,
                                                const float* __restrict__ bias,
                                                _Float16* __restrict__ qh,
                                                _Float16* __restrict__ kh,
                                                _Float16* __restrict__ vT) {
  __shared__ __align__(16) _Float16 As[64][64];   // x tile   [m][k]
  __shared__ __align__(16) _Float16 Bt[64][64];   // w tile   [j][k] (B transposed)
  const int m0 = blockIdx.x * 64;
  const int j0 = blockIdx.y * 64;
  const int tid = threadIdx.x;
  const int wave = tid >> 5;
  const int lane = tid & 31;
  const int s0 = wave * 2;      // two 16x16 subtiles per wave, same row band
  const int si = s0 >> 2;

  v8f acc0, acc1;
#pragma unroll
  for (int r = 0; r < 8; ++r) { acc0[r] = 0.f; acc1[r] = 0.f; }

  for (int k0 = 0; k0 < CX; k0 += 64) {
    __syncthreads();
#pragma unroll
    for (int i = tid; i < 64 * 64 / 8; i += 256) {   // 16B chunks
      int r = i >> 3, c8 = (i & 7) * 8;
      *(uint4*)&As[r][c8] = *(const uint4*)&xb[(long)(m0 + r) * CX + k0 + c8];
      *(uint4*)&Bt[r][c8] = *(const uint4*)&wb[(long)(j0 + r) * CX + k0 + c8];
    }
    __syncthreads();
#pragma unroll
    for (int kk = 0; kk < 64; kk += 32) {
      v16h a = frag_rm(&As[si * 16][kk], 64);
      v16h b0 = frag_rm(&Bt[(s0 & 3) * 16][kk], 64);
      v16h b1 = frag_rm(&Bt[((s0 + 1) & 3) * 16][kk], 64);
      acc0 = wmma32(a, b0, acc0);
      acc1 = wmma32(a, b1, acc1);
    }
  }

  const int colL = lane & 15, g = lane >> 4;
#pragma unroll
  for (int t = 0; t < 2; ++t) {
    v8f acc = t ? acc1 : acc0;
    const int sj = (s0 + t) & 3;
    const int j = j0 + sj * 16 + colL;
    const int which = j / CX;     // 0:q 1:k 2:v
    const int c = j % CX;
    const int h = c >> 6, d = c & 63;
    const float bj = bias[j];
#pragma unroll
    for (int r = 0; r < 8; ++r) {
      int m = m0 + si * 16 + r + 8 * g;
      int b = m >> 10, n = m & 1023;
      float val = acc[r] + bj;
      if (which == 0)
        qh[(((long)b * HX + h) * NX + n) * DX + d] = (_Float16)(val * SCALE_Q);
      else if (which == 1)
        kh[(((long)b * HX + h) * NX + n) * DX + d] = (_Float16)val;
      else  // v stored transposed [b][h][d][m] for row-major B tiles in attn@V
        vT[(((long)b * HX + h) * DX + d) * NX + n] = (_Float16)val;
    }
  }
}

// ---------------- fused softmax + head-mix + BN partial stats ----------------
// grid: (N/16, B), block: 128 (4 waves). Wave w owns 16-col subtile w of each
// 64-col m-tile, all 12 heads. Pre-BN mixed attn -> d_out attn region.
// K/Q tiles staged with global_load_async_to_lds_b128 (ASYNCcnt path).
__global__ __launch_bounds__(128) void attn_fused(const _Float16* __restrict__ qh,
                                                  const _Float16* __restrict__ kh,
                                                  const float* __restrict__ w_re,
                                                  const float* __restrict__ b_re,
                                                  float* __restrict__ attn_out,
                                                  float* __restrict__ stats) {
  extern __shared__ char smem[];
  _Float16* Qs = (_Float16*)smem;          // [12][16][64]  q rows (A operand)
  _Float16* Ks = Qs + HX * 16 * DX;        // [12][64][64]  k rows [m][d] (B^T)
  float* rowsum = (float*)(Ks + HX * 64 * DX);  // [12][16]
  float* wre = rowsum + HX * 16;           // [144]
  float* bre = wre + HX * HX;              // [12]

  const int tid = threadIdx.x;
  const int wave = tid >> 5;
  const int lane = tid & 31;
  const int n0 = blockIdx.x * 16;
  const int b = blockIdx.y;

  // ---- phase 0: stage Q (async b128), constants, zero row sums ----
  for (int i = tid; i < HX * 16 * DX / 8; i += 128) {   // 16B chunks
    int hr = i >> 3, c8 = (i & 7) * 8;
    int h = hr >> 4, row = hr & 15;
    async_copy_b128(&Qs[(h * 16 + row) * DX + c8],
                    &qh[(((long)b * HX + h) * NX + (n0 + row)) * DX + c8]);
  }
  for (int i = tid; i < HX * HX; i += 128) wre[i] = w_re[i];
  if (tid < HX) bre[tid] = b_re[tid];
  for (int i = tid; i < HX * 16; i += 128) rowsum[i] = 0.f;
  async_wait0();
  __syncthreads();

  // ---- phase 1: exp row sums (max-free softmax; logits are tiny) ----
  v8f sumf[HX];
#pragma unroll
  for (int h = 0; h < HX; ++h)
#pragma unroll
    for (int r = 0; r < 8; ++r) sumf[h][r] = 0.f;

  for (int mt = 0; mt < NX / 64; ++mt) {
    const int m0 = mt * 64;
    __syncthreads();
    for (int i = tid; i < HX * 64 * DX / 8; i += 128) {  // 16B chunks
      int hm = i >> 3, c8 = (i & 7) * 8;
      int h = hm >> 6, mr = hm & 63;
      async_copy_b128(&Ks[(h * 64 + mr) * DX + c8],
                      &kh[(((long)b * HX + h) * NX + (m0 + mr)) * DX + c8]);
    }
    async_wait0();
    __syncthreads();
#pragma unroll
    for (int h = 0; h < HX; ++h) {
      v8f acc;
#pragma unroll
      for (int r = 0; r < 8; ++r) acc[r] = 0.f;
#pragma unroll
      for (int kk = 0; kk < 64; kk += 32)
        acc = wmma32(frag_rm(Qs + h * 1024 + kk, DX),
                     frag_rm(Ks + h * 4096 + (wave * 16) * DX + kk, DX), acc);
#pragma unroll
      for (int r = 0; r < 8; ++r) sumf[h][r] += __expf(acc[r]);
    }
  }
  // cross-lane reduce within 16-lane halves, accumulate into LDS row sums
#pragma unroll
  for (int h = 0; h < HX; ++h) {
#pragma unroll
    for (int r = 0; r < 8; ++r) {
      float v = sumf[h][r];
      v += __shfl_xor(v, 1, 32);
      v += __shfl_xor(v, 2, 32);
      v += __shfl_xor(v, 4, 32);
      v += __shfl_xor(v, 8, 32);
      if ((lane & 15) == 0) atomicAdd(&rowsum[h * 16 + r + 8 * (lane >> 4)], v);
    }
  }
  __syncthreads();
  for (int i = tid; i < HX * 16; i += 128) rowsum[i] = 1.f / rowsum[i];
  __syncthreads();

  // ---- phase 2: recompute P, mix heads (12x12), write pre-BN, gather stats ----
  float ssum[HX], ssq[HX];
#pragma unroll
  for (int g2 = 0; g2 < HX; ++g2) { ssum[g2] = 0.f; ssq[g2] = 0.f; }

  for (int mt = 0; mt < NX / 64; ++mt) {
    const int m0 = mt * 64;
    __syncthreads();
    for (int i = tid; i < HX * 64 * DX / 8; i += 128) {
      int hm = i >> 3, c8 = (i & 7) * 8;
      int h = hm >> 6, mr = hm & 63;
      async_copy_b128(&Ks[(h * 64 + mr) * DX + c8],
                      &kh[(((long)b * HX + h) * NX + (m0 + mr)) * DX + c8]);
    }
    async_wait0();
    __syncthreads();

    v8f og[HX];
#pragma unroll
    for (int g2 = 0; g2 < HX; ++g2)
#pragma unroll
      for (int r = 0; r < 8; ++r) og[g2][r] = 0.f;

#pragma unroll
    for (int h = 0; h < HX; ++h) {
      v8f acc;
#pragma unroll
      for (int r = 0; r < 8; ++r) acc[r] = 0.f;
#pragma unroll
      for (int kk = 0; kk < 64; kk += 32)
        acc = wmma32(frag_rm(Qs + h * 1024 + kk, DX),
                     frag_rm(Ks + h * 4096 + (wave * 16) * DX + kk, DX), acc);
      v8f p;
#pragma unroll
      for (int r = 0; r < 8; ++r)
        p[r] = __expf(acc[r]) * rowsum[h * 16 + r + 8 * (lane >> 4)];
#pragma unroll
      for (int g2 = 0; g2 < HX; ++g2) {
        const float w = wre[g2 * HX + h];
#pragma unroll
        for (int r = 0; r < 8; ++r) og[g2][r] += w * p[r];
      }
    }
    const int colm = m0 + wave * 16 + (lane & 15);
#pragma unroll
    for (int g2 = 0; g2 < HX; ++g2) {
      const float bg = bre[g2];
#pragma unroll
      for (int r = 0; r < 8; ++r) {
        const int row = n0 + r + 8 * (lane >> 4);
        const float val = og[g2][r] + bg;
        attn_out[(((long)b * HX + g2) * NX + row) * NX + colm] = val;
        ssum[g2] += val;
        ssq[g2] += val * val;
      }
    }
  }
  // wave-wide stats reduce, one atomic per (wave, channel)
#pragma unroll
  for (int g2 = 0; g2 < HX; ++g2) {
    float s = ssum[g2], q = ssq[g2];
#pragma unroll
    for (int m = 1; m < 32; m <<= 1) {
      s += __shfl_xor(s, m, 32);
      q += __shfl_xor(q, m, 32);
    }
    if (lane == 0) {
      atomicAdd(&stats[g2], s);
      atomicAdd(&stats[HX + g2], q);
    }
  }
}

// ---------------- BN finalize: per-channel scale/shift ----------------
__global__ void bn_finalize(const float* __restrict__ stats,
                            const float* __restrict__ gamma,
                            const float* __restrict__ beta,
                            float* __restrict__ bnp) {
  int g = threadIdx.x;
  if (g < HX) {
    const float cnt = (float)((long)BX * NX * NX);
    float mean = stats[g] / cnt;
    float var = stats[HX + g] / cnt - mean * mean;
    float sc = gamma[g] * rsqrtf(var + BN_EPS);
    bnp[g] = sc;
    bnp[HX + g] = beta[g] - mean * sc;
  }
}

// ---------------- BN apply in place (float4) ----------------
__global__ __launch_bounds__(256) void bn_apply(float* __restrict__ attn,
                                                const float* __restrict__ bnp) {
  __shared__ float sc[HX], sh[HX];
  if (threadIdx.x < HX) {
    sc[threadIdx.x] = bnp[threadIdx.x];
    sh[threadIdx.x] = bnp[HX + threadIdx.x];
  }
  __syncthreads();
  const long total4 = ATTN_E / 4;
  const long stride = (long)gridDim.x * blockDim.x;
  float4* a4 = (float4*)attn;
  for (long i = (long)blockIdx.x * blockDim.x + threadIdx.x; i < total4; i += stride) {
    int g = (int)((i >> 18) % HX);   // (i*4)>>20 : 2^20 elems per channel slab
    float4 v = a4[i];
    v.x = v.x * sc[g] + sh[g];
    v.y = v.y * sc[g] + sh[g];
    v.z = v.z * sc[g] + sh[g];
    v.w = v.w * sc[g] + sh[g];
    a4[i] = v;
  }
}

// ---------------- attn @ V : per (b,h) [1024,1024] x [1024,64] ----------------
__global__ __launch_bounds__(128) void av_gemm(const float* __restrict__ attnN,
                                               const _Float16* __restrict__ vT,
                                               _Float16* __restrict__ out2) {
  __shared__ __align__(16) _Float16 As[64][64];   // attn tile [n][m] (f32->f16)
  __shared__ __align__(16) _Float16 Bt[64][64];   // vT tile   [d][m] (B^T)
  const int tid = threadIdx.x, wave = tid >> 5, lane = tid & 31;
  const int n0 = blockIdx.x * 64;
  const int bh = blockIdx.y;
  const int b = bh / HX, h = bh % HX;
  const float* Abase = attnN + ((long)b * HX + h) * NX * NX;
  const _Float16* Vbase = vT + ((long)b * HX + h) * DX * NX;

  v8f acc[4];
#pragma unroll
  for (int s = 0; s < 4; ++s)
#pragma unroll
    for (int r = 0; r < 8; ++r) acc[s][r] = 0.f;

  for (int k0 = 0; k0 < NX; k0 += 64) {
    __syncthreads();
#pragma unroll
    for (int i = tid; i < 64 * 64 / 4; i += 128) {   // f32 load + cvt, 8B store
      int r = i >> 4, c4 = (i & 15) * 4;
      float4 fv = *(const float4*)&Abase[(long)(n0 + r) * NX + k0 + c4];
      _Float16 h4[4] = {(_Float16)fv.x, (_Float16)fv.y,
                        (_Float16)fv.z, (_Float16)fv.w};
      *(uint2*)&As[r][c4] = *(uint2*)h4;
    }
#pragma unroll
    for (int i = tid; i < 64 * 64 / 8; i += 128) {   // 16B chunks
      int r = i >> 3, c8 = (i & 7) * 8;
      *(uint4*)&Bt[r][c8] = *(const uint4*)&Vbase[(long)r * NX + k0 + c8];
    }
    __syncthreads();
#pragma unroll
    for (int kk = 0; kk < 64; kk += 32) {
      v16h a = frag_rm(&As[wave * 16][kk], 64);
#pragma unroll
      for (int sj = 0; sj < 4; ++sj)
        acc[sj] = wmma32(a, frag_rm(&Bt[sj * 16][kk], 64), acc[sj]);
    }
  }
#pragma unroll
  for (int sj = 0; sj < 4; ++sj) {
    const int d = sj * 16 + (lane & 15);
#pragma unroll
    for (int r = 0; r < 8; ++r) {
      const int n = n0 + wave * 16 + r + 8 * (lane >> 4);
      out2[((long)b * NX + n) * CX + h * DX + d] = (_Float16)acc[sj][r];
    }
  }
}

// ---------------- proj GEMM: [8192,768] x [768,768]^T + bias -> f32 ----------------
__global__ __launch_bounds__(256) void proj_gemm(const _Float16* __restrict__ xb,
                                                 const _Float16* __restrict__ wb,
                                                 const float* __restrict__ bias,
                                                 float* __restrict__ out) {
  __shared__ __align__(16) _Float16 As[64][64];
  __shared__ __align__(16) _Float16 Bt[64][64];
  const int m0 = blockIdx.x * 64;
  const int j0 = blockIdx.y * 64;
  const int tid = threadIdx.x;
  const int wave = tid >> 5;
  const int lane = tid & 31;
  const int s0 = wave * 2;
  const int si = s0 >> 2;

  v8f acc0, acc1;
#pragma unroll
  for (int r = 0; r < 8; ++r) { acc0[r] = 0.f; acc1[r] = 0.f; }

  for (int k0 = 0; k0 < CX; k0 += 64) {
    __syncthreads();
#pragma unroll
    for (int i = tid; i < 64 * 64 / 8; i += 256) {
      int r = i >> 3, c8 = (i & 7) * 8;
      *(uint4*)&As[r][c8] = *(const uint4*)&xb[(long)(m0 + r) * CX + k0 + c8];
      *(uint4*)&Bt[r][c8] = *(const uint4*)&wb[(long)(j0 + r) * CX + k0 + c8];
    }
    __syncthreads();
#pragma unroll
    for (int kk = 0; kk < 64; kk += 32) {
      v16h a = frag_rm(&As[si * 16][kk], 64);
      v16h b0 = frag_rm(&Bt[(s0 & 3) * 16][kk], 64);
      v16h b1 = frag_rm(&Bt[((s0 + 1) & 3) * 16][kk], 64);
      acc0 = wmma32(a, b0, acc0);
      acc1 = wmma32(a, b1, acc1);
    }
  }

  const int colL = lane & 15, g = lane >> 4;
#pragma unroll
  for (int t = 0; t < 2; ++t) {
    v8f acc = t ? acc1 : acc0;
    const int sj = (s0 + t) & 3;
    const int j = j0 + sj * 16 + colL;
    const float bj = bias[j];
#pragma unroll
    for (int r = 0; r < 8; ++r) {
      int m = m0 + si * 16 + r + 8 * g;
      out[(long)m * CX + j] = acc[r] + bj;
    }
  }
}

// ---------------- host-side orchestration ----------------
extern "C" void kernel_launch(void* const* d_in, const int* in_sizes, int n_in,
                              void* d_out, int out_size, void* d_ws, size_t ws_size,
                              hipStream_t stream) {
  (void)in_sizes; (void)n_in; (void)out_size; (void)ws_size;
  const float* x        = (const float*)d_in[0];
  const float* w_qkv    = (const float*)d_in[1];
  const float* b_qkv    = (const float*)d_in[2];
  const float* w_re     = (const float*)d_in[3];
  const float* b_re     = (const float*)d_in[4];
  const float* bn_gamma = (const float*)d_in[5];
  const float* bn_beta  = (const float*)d_in[6];
  const float* w_proj   = (const float*)d_in[7];
  const float* b_proj   = (const float*)d_in[8];

  char* ws = (char*)d_ws;
  _Float16* xb     = (_Float16*)ws; ws += XB_E * sizeof(_Float16);
  _Float16* wqkvb  = (_Float16*)ws; ws += WQKV_E * sizeof(_Float16);
  _Float16* wprojb = (_Float16*)ws; ws += WPROJ_E * sizeof(_Float16);
  _Float16* qh     = (_Float16*)ws; ws += HEAD_E * sizeof(_Float16);
  _Float16* kh     = (_Float16*)ws; ws += HEAD_E * sizeof(_Float16);
  _Float16* vT     = (_Float16*)ws; ws += HEAD_E * sizeof(_Float16);
  _Float16* out2   = (_Float16*)ws; ws += OUT_E * sizeof(_Float16);
  float* stats     = (float*)ws;    ws += 2 * HX * sizeof(float);
  float* bnp       = (float*)ws;    ws += 2 * HX * sizeof(float);

  float* outp = (float*)d_out;
  float* attn_out = outp + OUT_E;

  hipMemsetAsync(stats, 0, 2 * HX * sizeof(float), stream);

  cvt_f32_f16<<<(unsigned)((XB_E + 255) / 256), 256, 0, stream>>>(x, xb, XB_E);
  cvt_f32_f16<<<(unsigned)((WQKV_E + 255) / 256), 256, 0, stream>>>(w_qkv, wqkvb, WQKV_E);
  cvt_f32_f16<<<(unsigned)((WPROJ_E + 255) / 256), 256, 0, stream>>>(w_proj, wprojb, WPROJ_E);

  qkv_gemm<<<dim3(BX * NX / 64, 3 * CX / 64), 256, 0, stream>>>(
      xb, wqkvb, b_qkv, qh, kh, vT);

  const size_t attn_lds = (size_t)(HX * 16 * DX + HX * 64 * DX) * sizeof(_Float16) +
                          (size_t)(HX * 16 + HX * HX + HX) * sizeof(float);
  attn_fused<<<dim3(NX / 16, BX), 128, attn_lds, stream>>>(
      qh, kh, w_re, b_re, attn_out, stats);

  bn_finalize<<<1, 32, 0, stream>>>(stats, bn_gamma, bn_beta, bnp);
  bn_apply<<<4096, 256, 0, stream>>>(attn_out, bnp);

  av_gemm<<<dim3(NX / 64, BX * HX), 128, 0, stream>>>(attn_out, vT, out2);

  proj_gemm<<<dim3(BX * NX / 64, CX / 64), 256, 0, stream>>>(
      out2, wprojb, b_proj, outp);
}
